// MoDAAttention_31585189495117
// MI455X (gfx1250) — matrix operand order repaired and checked by reading, not compiled, not run on hardware
//
#include <hip/hip_runtime.h>
#include <math.h>
#include <stdint.h>

#define B_  2
#define T_  2048
#define D_  1024
#define HQ_ 16
#define HK_ 4
#define HD_ 64
#define LP_ 8
#define GQ_ (HQ_/HK_)

typedef __bf16 bf16;
typedef __attribute__((ext_vector_type(16))) bf16  v16bf;
typedef __attribute__((ext_vector_type(8)))  bf16  v8bf;
typedef __attribute__((ext_vector_type(8)))  float v8f;
typedef __attribute__((ext_vector_type(4)))  unsigned int u32x4;
typedef __attribute__((ext_vector_type(8)))  int i32x8;
typedef __attribute__((ext_vector_type(4)))  int i32x4;

static __device__ __forceinline__ v16bf make_v16(v8bf lo, v8bf hi) {
  v16bf r;
#pragma unroll
  for (int i = 0; i < 8; ++i) { r[i] = lo[i]; r[i + 8] = hi[i]; }
  return r;
}

static __device__ __forceinline__ v8f wmma_bf16(v16bf a, v16bf b, v8f c) {
  return __builtin_amdgcn_wmma_f32_16x16x32_bf16(false, a, false, b, (short)0, c,
                                                 false, false);
}

// ---------------------------------------------------------------- converts
__global__ void cvt_f32_to_bf16(const float* __restrict__ in,
                                bf16* __restrict__ out, long n) {
  long i = (long)blockIdx.x * blockDim.x + threadIdx.x;
  long stride = (long)gridDim.x * blockDim.x;
  for (; i < n; i += stride) out[i] = (bf16)in[i];
}

// ---------------------------------------------------------------- GEMM
// C[M,N] = A(MxK, bf16 row-major) @ W(KxN, bf16 row-major), f32 accum.
// Block = 4 waves, 64x64 tile. W K-slab staged transposed in LDS.
__global__ void __launch_bounds__(128)
gemm_bf16_wmma(const bf16* __restrict__ A, const bf16* __restrict__ W,
               int M, int N, int K,
               bf16* __restrict__ outb, float* __restrict__ outf,
               int head_major, int nheads) {
  __shared__ __align__(64) bf16 ldsW[64 * 32];  // [n_local][k_local]
  const int tid  = threadIdx.x;
  const int wave = tid >> 5;
  const int lane = tid & 31;
  const int hh   = lane >> 4;   // lane half: selects K sub-range / M+8
  const int l16  = lane & 15;
  const int m0   = blockIdx.x * 64 + wave * 16;
  const int n0   = blockIdx.y * 64;

  v8f acc[4] = {};

  const int kl  = tid >> 2;         // 0..31: k row this thread stages
  const int ns  = (tid & 3) << 4;   // 0/16/32/48: n segment
  const int arow = m0 + l16;
  const int klo  = hh * 16;

  for (int k0 = 0; k0 < K; k0 += 32) {
    const bf16* wp = W + (size_t)(k0 + kl) * N + (n0 + ns);
    v8bf w0 = *(const v8bf*)wp;
    v8bf w1 = *(const v8bf*)(wp + 8);
#pragma unroll
    for (int i = 0; i < 8; ++i) {
      ldsW[(ns + i)     * 32 + kl] = w0[i];
      ldsW[(ns + 8 + i) * 32 + kl] = w1[i];
    }
    if (k0 + 32 < K) __builtin_prefetch(wp + (size_t)32 * N, 0, 0);
    __syncthreads();

    const bf16* ap = A + (size_t)arow * K + k0;
    v16bf afrag = make_v16(*(const v8bf*)(ap + klo),
                           *(const v8bf*)(ap + 8 + klo));
#pragma unroll
    for (int sub = 0; sub < 4; ++sub) {
      const bf16* bp = &ldsW[(sub * 16 + l16) * 32 + hh * 16];
      acc[sub] = wmma_bf16(afrag, *(const v16bf*)bp, acc[sub]);
    }
    __syncthreads();
  }

#pragma unroll
  for (int sub = 0; sub < 4; ++sub) {
#pragma unroll
    for (int r = 0; r < 8; ++r) {
      int m = m0 + r + hh * 8;
      int n = n0 + sub * 16 + l16;
      float v = acc[sub][r];
      size_t idx;
      if (head_major) {
        int bb = m / T_, tt = m % T_;
        int h = n >> 6, hd = n & 63;
        idx = ((size_t)(bb * nheads + h) * T_ + tt) * HD_ + hd;
      } else {
        idx = (size_t)m * N + n;
      }
      if (outb) outb[idx] = (bf16)v;
      if (outf) outf[idx] = v;
    }
  }
}

// ---------------------------------------------------------------- attention
// grid = (T/64, B*HQ), block = 128 (4 waves). Each wave: 16 queries.
// K tile DMA'd by the TDM (wave 0), V tile async-copied to LDS then
// LDS-transposed. Flash loop over key pairs (32 keys), then LP depth slots.
__global__ void __launch_bounds__(128)
moda_attn_wmma(const bf16* __restrict__ Qb, const bf16* __restrict__ Kb,
               const bf16* __restrict__ Vb, const bf16* __restrict__ Kdb,
               const bf16* __restrict__ Vdb, bf16* __restrict__ attnO) {
  const float scale = 0.125f;  // 64^-0.5
  const int bh  = blockIdx.y;
  const int b   = bh / HQ_;
  const int h   = bh % HQ_;
  const int bk  = b * HK_ + (h / GQ_);
  const int tid = threadIdx.x, wave = tid >> 5, lane = tid & 31;
  const int hh  = lane >> 4, l16 = lane & 15;
  const int q0b = blockIdx.x * 64;
  const int q0w = q0b + wave * 16;

  __shared__ __align__(64) bf16  ldsK[32 * 64];    // [key][d]   (TDM target)
  __shared__ __align__(64) bf16  ldsVrm[32 * 64];  // [key][d]   (async target)
  __shared__ __align__(64) bf16  ldsVt[64 * 32];   // [d][key]   (transposed)
  __shared__ __align__(64) bf16  ldsP[4][16 * 32]; // per-wave P tile
  __shared__ float ldsDep[4][16 * LP_];

  // Q fragments (16x64 row-major tile)
  const bf16* qbase = Qb + ((size_t)bh * T_ + q0w) * HD_;
  const int   klo   = hh * 16;
  const bf16* qp    = qbase + (size_t)l16 * HD_;
  v16bf qa0 = make_v16(*(const v8bf*)(qp + klo),      *(const v8bf*)(qp + 8 + klo));
  v16bf qa1 = make_v16(*(const v8bf*)(qp + 32 + klo), *(const v8bf*)(qp + 40 + klo));

  float mrow[8], Lrow[8];
#pragma unroll
  for (int r = 0; r < 8; ++r) { mrow[r] = -3.0e38f; Lrow[r] = 0.0f; }
  v8f O[4] = {};

  const int key_l = tid >> 2;         // 0..31: key row this thread stages
  const int dseg  = (tid & 3) << 4;   // 0/16/32/48
  const int npairs = (q0b >> 5) + 2;  // cover keys up to q0b+63

  const bf16* Kh = Kb + (size_t)bk * T_ * HD_;
  const bf16* Vh = Vb + (size_t)bk * T_ * HD_;

  const unsigned ldsK_off  = (unsigned)(uintptr_t)(&ldsK[0]);
  const unsigned ldsV_lane = (unsigned)(uintptr_t)(&ldsVrm[key_l * 64 + dseg]);

  for (int jj = 0; jj < npairs; ++jj) {
    const int k0 = jj * 32;

    // --- K tile via Tensor Data Mover (one descriptor per block) ---
    if (wave == 0) {
      unsigned long long ga =
          (unsigned long long)(uintptr_t)(Kh + (size_t)k0 * HD_);
      u32x4 g0;
      g0[0] = 1u;                         // count=1, user descriptor
      g0[1] = ldsK_off;                   // lds_addr
      g0[2] = (unsigned)ga;               // global_addr[31:0]
      g0[3] = ((unsigned)(ga >> 32) & 0x01FFFFFFu) | (2u << 30);  // addr hi, type=2
      i32x8 g1;
      g1[0] = 0x10000;        // workgroup_mask=0, data_size=1 (2B)
      g1[1] = (HD_ << 16);    // tensor_dim0[15:0] = 64
      g1[2] = (T_ << 16);     // tensor_dim0[31:16]=0 | tensor_dim1[15:0]=2048
      g1[3] = (HD_ << 16);    // tensor_dim1[31:16]=0 | tile_dim0 = 64
      g1[4] = 32;             // tile_dim1 = 32, tile_dim2 = 0
      g1[5] = HD_;            // tensor_dim0_stride[31:0] = 64
      g1[6] = 0;
      g1[7] = 0;
      i32x4 gz4 = {0, 0, 0, 0};
      i32x8 gz8 = {0, 0, 0, 0, 0, 0, 0, 0};
      __builtin_amdgcn_tensor_load_to_lds(g0, g1, gz4, gz4, gz8, 0);
      __builtin_amdgcn_s_wait_tensorcnt(0);
    }

    // --- V tile rows via async global->LDS copy (row-major staging) ---
    {
      unsigned long long vga =
          (unsigned long long)(uintptr_t)(Vh + (size_t)(k0 + key_l) * HD_ + dseg);
      asm volatile("global_load_async_to_lds_b128 %0, %1, off"
                   :: "v"(ldsV_lane), "v"(vga) : "memory");
      asm volatile("global_load_async_to_lds_b128 %0, %1, off offset:16"
                   :: "v"(ldsV_lane), "v"(vga) : "memory");
      asm volatile("s_wait_asynccnt 0x0" ::: "memory");
    }
    __syncthreads();

    // --- LDS transpose of V: [key][d] -> [d][key] ---
    {
      v8bf vv0 = *(const v8bf*)&ldsVrm[key_l * 64 + dseg];
      v8bf vv1 = *(const v8bf*)&ldsVrm[key_l * 64 + dseg + 8];
#pragma unroll
      for (int i = 0; i < 8; ++i) {
        ldsVt[(dseg + i)     * 32 + key_l] = vv0[i];
        ldsVt[(dseg + 8 + i) * 32 + key_l] = vv1[i];
      }
    }

    // S = scale * Q @ K^T (two 16x16 tiles), causal-masked
    v8f s[2];
#pragma unroll
    for (int t2 = 0; t2 < 2; ++t2) {
      const bf16* kb0 = &ldsK[(t2 * 16 + l16) * 64 + 0  + hh * 16];
      const bf16* kb1 = &ldsK[(t2 * 16 + l16) * 64 + 32 + hh * 16];
      v8f c = {};
      c = wmma_bf16(qa0, *(const v16bf*)kb0, c);
      c = wmma_bf16(qa1, *(const v16bf*)kb1, c);
      const int key = k0 + t2 * 16 + l16;
#pragma unroll
      for (int r = 0; r < 8; ++r) {
        int q = q0w + r + hh * 8;
        s[t2][r] = (key <= q) ? c[r] * scale : -1.0e30f;
      }
    }

    // online softmax (row reductions across 16-lane groups)
    float alpha[8];
#pragma unroll
    for (int r = 0; r < 8; ++r) {
      float tmax = fmaxf(s[0][r], s[1][r]);
#pragma unroll
      for (int off = 8; off >= 1; off >>= 1)
        tmax = fmaxf(tmax, __shfl_xor(tmax, off, 16));
      float mnew = fmaxf(mrow[r], tmax);
      alpha[r] = __expf(mrow[r] - mnew);
      float p0 = __expf(s[0][r] - mnew);
      float p1 = __expf(s[1][r] - mnew);
      float rs = p0 + p1;
#pragma unroll
      for (int off = 8; off >= 1; off >>= 1)
        rs += __shfl_xor(rs, off, 16);
      Lrow[r] = Lrow[r] * alpha[r] + rs;
      mrow[r] = mnew;
      ldsP[wave][(r + hh * 8) * 32 + l16]      = (bf16)p0;
      ldsP[wave][(r + hh * 8) * 32 + 16 + l16] = (bf16)p1;
    }
#pragma unroll
    for (int sub = 0; sub < 4; ++sub)
#pragma unroll
      for (int r = 0; r < 8; ++r) O[sub][r] *= alpha[r];
    __syncthreads();   // ldsP + ldsVt ready for all waves

    // O += P(16x32) @ V(32x64)
    const bf16* pp = &ldsP[wave][l16 * 32];
    v16bf pa = make_v16(*(const v8bf*)(pp + klo), *(const v8bf*)(pp + 8 + klo));
#pragma unroll
    for (int sub = 0; sub < 4; ++sub) {
      const bf16* vbp = &ldsVt[(sub * 16 + l16) * 32 + hh * 16];
      O[sub] = wmma_bf16(pa, *(const v16bf*)vbp, O[sub]);
    }
    __syncthreads();   // protect next iteration's tile overwrite
  }

  // --- depth slots: s_dep[t,l] = scale * Q[t] . Kd[t,l] ---
  {
    const size_t tbase = (size_t)bk * T_;
#pragma unroll
    for (int i = 0; i < 4; ++i) {
      int p  = lane + 32 * i;   // 0..127 -> (row, slot)
      int rr = p >> 3;
      int sl = p & 7;
      int tg = q0w + rr;
      const bf16* qr = qbase + (size_t)rr * HD_;
      const bf16* kd = Kdb + ((tbase + tg) * LP_ + sl) * HD_;
      float acc = 0.f;
#pragma unroll
      for (int d = 0; d < HD_; ++d) acc += (float)qr[d] * (float)kd[d];
      ldsDep[wave][rr * LP_ + sl] = acc * scale;
    }
  }
  __syncthreads();

  // fold depth into softmax + accumulate depth value contribution
  {
    const size_t tbase = (size_t)bk * T_;
#pragma unroll
    for (int r = 0; r < 8; ++r) {
      int rowloc = r + hh * 8;
      int tg = q0w + rowloc;
      float dl[LP_], dmax = -3.0e38f;
#pragma unroll
      for (int l = 0; l < LP_; ++l) {
        dl[l] = ldsDep[wave][rowloc * LP_ + l];
        dmax = fmaxf(dmax, dl[l]);
      }
      float mnew  = fmaxf(mrow[r], dmax);
      float alpha = __expf(mrow[r] - mnew);
      float pd[LP_], ds = 0.f;
#pragma unroll
      for (int l = 0; l < LP_; ++l) { pd[l] = __expf(dl[l] - mnew); ds += pd[l]; }
      Lrow[r] = Lrow[r] * alpha + ds;
      mrow[r] = mnew;
      const bf16* vdb = Vdb + ((tbase + tg) * LP_) * HD_;
#pragma unroll
      for (int sub = 0; sub < 4; ++sub) {
        int col = sub * 16 + l16;
        float acc = O[sub][r] * alpha;
#pragma unroll
        for (int l = 0; l < LP_; ++l) acc += pd[l] * (float)vdb[l * HD_ + col];
        O[sub][r] = acc;
      }
    }
  }

  // normalize + store (b, t, hq*hd) bf16 for the output projection
#pragma unroll
  for (int r = 0; r < 8; ++r) {
    int rowloc = r + hh * 8;
    int tg = q0w + rowloc;
    float inv = 1.0f / Lrow[r];
    bf16* op = attnO + ((size_t)b * T_ + tg) * (HQ_ * HD_) + h * HD_;
#pragma unroll
    for (int sub = 0; sub < 4; ++sub)
      op[sub * 16 + l16] = (bf16)(O[sub][r] * inv);
  }
}

// ---------------------------------------------------------------- launch
extern "C" void kernel_launch(void* const* d_in, const int* in_sizes, int n_in,
                              void* d_out, int out_size, void* d_ws, size_t ws_size,
                              hipStream_t stream) {
  (void)in_sizes; (void)n_in; (void)out_size; (void)ws_size;
  const float* x     = (const float*)d_in[0];
  const float* K_dep = (const float*)d_in[1];
  const float* V_dep = (const float*)d_in[2];
  const float* Wq    = (const float*)d_in[3];
  const float* Wk    = (const float*)d_in[4];
  const float* Wv    = (const float*)d_in[5];
  const float* Wkd   = (const float*)d_in[6];
  const float* Wvd   = (const float*)d_in[7];
  const float* Wo    = (const float*)d_in[8];

  float* out = (float*)d_out;
  float* kw  = out + (size_t)B_ * T_ * D_;
  float* vw  = kw + (size_t)B_ * HK_ * T_ * HD_;

  char* p = (char*)d_ws;
  auto take = [&](size_t elems) { bf16* r = (bf16*)p; p += elems * sizeof(bf16); return r; };
  bf16* xb   = take((size_t)B_ * T_ * D_);
  bf16* Wqb  = take((size_t)D_ * HQ_ * HD_);
  bf16* Wkb  = take((size_t)D_ * HK_ * HD_);
  bf16* Wvb  = take((size_t)D_ * HK_ * HD_);
  bf16* Wkdb = take((size_t)D_ * HK_ * HD_);
  bf16* Wvdb = take((size_t)D_ * HK_ * HD_);
  bf16* Wob  = take((size_t)D_ * D_);
  bf16* Qb   = take((size_t)B_ * HQ_ * T_ * HD_);
  bf16* Kb   = take((size_t)B_ * HK_ * T_ * HD_);
  bf16* Vb   = take((size_t)B_ * HK_ * T_ * HD_);
  bf16* Kdb  = take((size_t)B_ * HK_ * T_ * LP_ * HD_);
  bf16* Vdb  = take((size_t)B_ * HK_ * T_ * LP_ * HD_);
  bf16* aO   = take((size_t)B_ * T_ * HQ_ * HD_);

  dim3 cb(256), cg(1024);
  cvt_f32_to_bf16<<<cg, cb, 0, stream>>>(x,     xb,   (long)B_ * T_ * D_);
  cvt_f32_to_bf16<<<cg, cb, 0, stream>>>(Wq,    Wqb,  (long)D_ * HQ_ * HD_);
  cvt_f32_to_bf16<<<cg, cb, 0, stream>>>(Wk,    Wkb,  (long)D_ * HK_ * HD_);
  cvt_f32_to_bf16<<<cg, cb, 0, stream>>>(Wv,    Wvb,  (long)D_ * HK_ * HD_);
  cvt_f32_to_bf16<<<cg, cb, 0, stream>>>(Wkd,   Wkdb, (long)D_ * HK_ * HD_);
  cvt_f32_to_bf16<<<cg, cb, 0, stream>>>(Wvd,   Wvdb, (long)D_ * HK_ * HD_);
  cvt_f32_to_bf16<<<cg, cb, 0, stream>>>(Wo,    Wob,  (long)D_ * D_);
  cvt_f32_to_bf16<<<cg, cb, 0, stream>>>(K_dep, Kdb,  (long)B_ * HK_ * T_ * LP_ * HD_);
  cvt_f32_to_bf16<<<cg, cb, 0, stream>>>(V_dep, Vdb,  (long)B_ * HK_ * T_ * LP_ * HD_);

  const int M = B_ * T_;
  dim3 gb(128);
  gemm_bf16_wmma<<<dim3(M / 64, (HQ_ * HD_) / 64), gb, 0, stream>>>(
      xb, Wqb, M, HQ_ * HD_, D_, Qb, nullptr, 1, HQ_);
  gemm_bf16_wmma<<<dim3(M / 64, (HK_ * HD_) / 64), gb, 0, stream>>>(
      xb, Wkb, M, HK_ * HD_, D_, Kb, nullptr, 1, HK_);
  gemm_bf16_wmma<<<dim3(M / 64, (HK_ * HD_) / 64), gb, 0, stream>>>(
      xb, Wvb, M, HK_ * HD_, D_, Vb, nullptr, 1, HK_);
  gemm_bf16_wmma<<<dim3(M / 64, (HK_ * HD_) / 64), gb, 0, stream>>>(
      xb, Wkdb, M, HK_ * HD_, D_, nullptr, kw, 1, HK_);
  gemm_bf16_wmma<<<dim3(M / 64, (HK_ * HD_) / 64), gb, 0, stream>>>(
      xb, Wvdb, M, HK_ * HD_, D_, nullptr, vw, 1, HK_);

  moda_attn_wmma<<<dim3(T_ / 64, B_ * HQ_), gb, 0, stream>>>(Qb, Kb, Vb, Kdb, Vdb, aO);

  gemm_bf16_wmma<<<dim3(M / 64, D_ / 64), gb, 0, stream>>>(
      aO, Wob, M, D_, D_, nullptr, out, 0, 0);
}
